// OuterProducterMean_10685878633040
// MI455X (gfx1250) — compile-verified, matrix operand order checked by reading, not run on hardware
//
#include <hip/hip_runtime.h>

// OuterProductMean fused WMMA implementation for gfx1250 (MI455X).
// N=1, S=128, R=256, M=256, C=32, CZ=128
//
// Workspace layout (needs 4 MB):
//   Lbuf bf16 [R*C][S]  (row = r*32+c, col = s)   2 MB
//   Rbuf bf16 [R*C][S]  (row = t*32+e, col = s)   2 MB

#define S_DIM 128
#define R_DIM 256
#define M_DIM 256
#define C_DIM 32
#define CZ_DIM 128

typedef __attribute__((ext_vector_type(16))) __bf16 v16bf;
typedef __attribute__((ext_vector_type(8)))  float  v8f;

// ---------------------------------------------------------------------------
// Operand loader following the documented CDNA5 16-bit WMMA VGPR layouts.
// A (16x32, MxK) from row-major [row][k] storage:
//   lane -> M = lane&15; VGPR v pair holds
//   K = (v>=4?16:0) + (lane>=16?8:0) + (v%4)*2 + {0,1}
// The same pattern serves the B operand (32x16, KxN) when B is stored
// n-major as [n][k] (lane -> N = lane&15).
// Per lane this is two contiguous 16-byte runs -> two ds_load_b128.
// ---------------------------------------------------------------------------
__device__ inline v16bf wmma_load_rowmajor(const __bf16* __restrict__ base,
                                           int stride, int m0, int kOff,
                                           int lane) {
  int mrow = lane & 15;
  int half = lane >> 4;
  const __bf16* p = base + (m0 + mrow) * stride + kOff;
  v16bf a;
#pragma unroll
  for (int v = 0; v < 8; ++v) {
    int kb = ((v & 4) << 2) + half * 8 + ((v & 3) << 1);
    a[2 * v]     = p[kb];
    a[2 * v + 1] = p[kb + 1];
  }
  return a;
}

// Async global -> LDS copy of one 16-byte chunk (CDNA5 GLOBAL_LOAD_ASYNC,
// tracked by ASYNCcnt; bypasses VGPRs). lds_addr is the per-lane LDS byte
// address (low 32 bits of the generic pointer), gaddr the global address.
__device__ inline void async_copy_b128(unsigned lds_addr,
                                       const void* __restrict__ gptr) {
  unsigned long long ga = (unsigned long long)(uintptr_t)gptr;
  asm volatile("global_load_async_to_lds_b128 %0, %1, off"
               :: "v"(lds_addr), "v"(ga)
               : "memory");
}

__device__ inline void wait_asynccnt0() {
  asm volatile("s_wait_asynccnt 0x0" ::: "memory");
}

// ---------------------------------------------------------------------------
// Kernel 1: projections  left/right[s,r,c] = sum_m msa[s,r,m] * w[m,c]
// Writes bf16, transposed to [(r,c)][s] so stage-2 WMMA K (=s) is contiguous.
// 1.1 GFLOP total; memory-latency bound; plain VALU is fine here.
// ---------------------------------------------------------------------------
__global__ __launch_bounds__(64) void proj_kernel(
    const float* __restrict__ msa, const float* __restrict__ lw,
    const float* __restrict__ rw, __bf16* __restrict__ Lbuf,
    __bf16* __restrict__ Rbuf) {
  __shared__ float row[M_DIM];
  int blk = blockIdx.x;  // blk = s*R + r
  int s = blk / R_DIM;
  int r = blk % R_DIM;
  const float* src = msa + (size_t)blk * M_DIM;
  for (int i = threadIdx.x; i < M_DIM; i += 64) row[i] = src[i];
  __syncthreads();
  int c = threadIdx.x & 31;
  int isR = threadIdx.x >> 5;
  const float* w = isR ? rw : lw;
  float acc = 0.f;
#pragma unroll 8
  for (int m = 0; m < M_DIM; ++m) acc += row[m] * w[m * C_DIM + c];
  __bf16* dst = isR ? Rbuf : Lbuf;
  dst[((size_t)r * C_DIM + c) * S_DIM + s] = (__bf16)acc;
}

// ---------------------------------------------------------------------------
// Kernel 2: fused outer-product + output projection.
// Block (bx,by) computes out[r0:r0+16, t0:t0+16, 0:128], r0=16*by, t0=16*bx.
// Per c in 0..31:
//   stage 2: outer[(r),(t,e)] = sum_s L[(r,c),s] * R[(t,e),s]
//            (GEMM 16 x 512 x 128 via WMMA bf16, result -> bf16 Osh[rt][e])
//   stage 3: out[(r,t), z] += Osh[(r,t), e] @ W[(c,e), z]
//            (GEMM 256 x 128 x 32 via WMMA bf16, f32 accumulators)
// LDS: 128K (R tile) + 4K (L slice) + 8K (W slice) + 16K (Osh) = 156 KB.
// ---------------------------------------------------------------------------
__global__ __launch_bounds__(256, 1) void opm_fused_kernel(
    const __bf16* __restrict__ Lbuf, const __bf16* __restrict__ Rbuf,
    const float* __restrict__ out_w, float* __restrict__ out) {
  __shared__ __bf16 Rsh[512][S_DIM];      // 128 KB: rows (tloc*32+e), col s
  __shared__ __bf16 Lc[16][S_DIM];        //   4 KB: rows rloc (fixed c), col s
  __shared__ __bf16 Wsh[CZ_DIM][C_DIM];   //   8 KB: TRANSPOSED [z][e] (n-major)
  __shared__ __bf16 Osh[256][C_DIM];      //  16 KB: rows rt=(rloc*16+tloc), col e

  const int tid  = threadIdx.x;
  const int lane = tid & 31;
  const int wv   = tid >> 5;        // 8 waves
  const int t0   = blockIdx.x * 16;
  const int r0   = blockIdx.y * 16;

  // ---- stage R tile once via async global->LDS (128 KB, 8192 x b128) ----
  {
    unsigned lds_base = (unsigned)(uintptr_t)&Rsh[0][0];
    const char* gsrc = (const char*)(Rbuf + (size_t)(t0 * C_DIM) * S_DIM);
#pragma unroll 4
    for (int i = tid; i < 512 * S_DIM * 2 / 16; i += 256)
      async_copy_b128(lds_base + (unsigned)i * 16, gsrc + (size_t)i * 16);
  }

  // persistent f32 accumulators: 2 rt-row-tiles x 8 z-tiles per wave
  v8f acc3[2][8];
#pragma unroll
  for (int q = 0; q < 2; ++q)
#pragma unroll
    for (int z = 0; z < 8; ++z) acc3[q][z] = 0.f;

  for (int c = 0; c < C_DIM; ++c) {
    __syncthreads();  // protect Lc/Wsh/Osh WAR vs previous iteration

    // ---- async-load L slice for this c: rows (r0+rloc)*32 + c ----
    {
      int rr = tid >> 4, chunk = tid & 15;  // 16 rows x 16 x b128 each
      unsigned lds_addr =
          (unsigned)(uintptr_t)&Lc[rr][0] + (unsigned)chunk * 16;
      const char* ga =
          (const char*)(Lbuf + ((size_t)((r0 + rr) * C_DIM + c)) * S_DIM) +
          chunk * 16;
      async_copy_b128(lds_addr, ga);
    }
    // ---- load + convert + transpose W slice: Wsh[z][e] = W[c*32+e][z] ----
    for (int i = tid; i < C_DIM * CZ_DIM; i += 256) {
      int e = i >> 7, z = i & 127;  // coalesced along z per global row
      Wsh[z][e] = (__bf16)out_w[(size_t)(c * C_DIM + e) * CZ_DIM + z];
    }
    // prefetch next iteration's W slice into cache while we compute
    if (c + 1 < C_DIM)
      __builtin_prefetch(out_w + (size_t)((c + 1) * C_DIM) * CZ_DIM, 0, 0);

    wait_asynccnt0();  // L slice resident (and R tile on first iteration)
    __syncthreads();

    // ---- stage 2: 32 n-tiles over (t,e); 4 per wave; K = s = 128 ----
#pragma unroll
    for (int nt = 0; nt < 4; ++nt) {
      int n0 = (wv * 4 + nt) * 16;          // col tile in (t,e) space
      int tloc = n0 >> 5;                   // tile never crosses t (32 e per t)
      int e0 = n0 & 31;
      v8f acc = 0.f;
#pragma unroll
      for (int kb = 0; kb < 4; ++kb) {
        v16bf a = wmma_load_rowmajor(&Lc[0][0], S_DIM, 0, kb * 32, lane);
        v16bf b = wmma_load_rowmajor(&Rsh[0][0], S_DIM, n0, kb * 32, lane);
        acc = __builtin_amdgcn_wmma_f32_16x16x32_bf16(
            false, a, false, b, (short)0, acc, false, false);
      }
      // D layout: VGPR i holds M = i + 8*(lane>=16), N = n0 + (lane&15)
      int ncol = lane & 15, half = lane >> 4;
#pragma unroll
      for (int i = 0; i < 8; ++i) {
        int rloc = i + 8 * half;
        Osh[rloc * 16 + tloc][e0 + ncol] = (__bf16)acc[i];
      }
    }
    __syncthreads();

    // ---- stage 3: rows rt (256) x cols z (128), K = e = 32 ----
#pragma unroll
    for (int q = 0; q < 2; ++q) {
      int rt0 = (wv * 2 + q) * 16;
      v16bf a = wmma_load_rowmajor(&Osh[0][0], C_DIM, rt0, 0, lane);
#pragma unroll
      for (int zt = 0; zt < 8; ++zt) {
        // B = W[e][z] tile; n-major storage Wsh[z][e] -> same loader as A
        v16bf b = wmma_load_rowmajor(&Wsh[0][0], C_DIM, zt * 16, 0, lane);
        acc3[q][zt] = __builtin_amdgcn_wmma_f32_16x16x32_bf16(
            false, a, false, b, (short)0, acc3[q][zt], false, false);
      }
    }
  }

  // ---- epilogue: write f32 output ----
  {
    int ncol = lane & 15, half = lane >> 4;
#pragma unroll
    for (int q = 0; q < 2; ++q) {
      int rt_base = (wv * 2 + q) * 16;
#pragma unroll
      for (int zt = 0; zt < 8; ++zt) {
#pragma unroll
        for (int i = 0; i < 8; ++i) {
          int rt = rt_base + i + 8 * half;
          int rloc = rt >> 4, tloc = rt & 15;
          out[((size_t)(r0 + rloc) * R_DIM + (t0 + tloc)) * CZ_DIM + zt * 16 + ncol] =
              acc3[q][zt][i];
        }
      }
    }
  }
}

// ---------------------------------------------------------------------------
extern "C" void kernel_launch(void* const* d_in, const int* in_sizes, int n_in,
                              void* d_out, int out_size, void* d_ws, size_t ws_size,
                              hipStream_t stream) {
  const float* msa = (const float*)d_in[0];  // [1,128,256,256]
  const float* lw  = (const float*)d_in[1];  // [256,32]
  const float* rw  = (const float*)d_in[2];  // [256,32]
  const float* ow  = (const float*)d_in[3];  // [1024,128]
  float* out = (float*)d_out;                // [256,256,128]

  __bf16* Lbuf = (__bf16*)d_ws;
  __bf16* Rbuf = Lbuf + (size_t)R_DIM * C_DIM * S_DIM;  // +1M elems (2 MB)

  proj_kernel<<<dim3(S_DIM * R_DIM), 64, 0, stream>>>(msa, lw, rw, Lbuf, Rbuf);
  opm_fused_kernel<<<dim3(R_DIM / 16, R_DIM / 16), 256, 0, stream>>>(
      Lbuf, Rbuf, ow, out);
}
